// Seq2SeqSemanticParserAttention_10514079941506
// MI455X (gfx1250) — matrix operand order repaired and checked by reading, not compiled, not run on hardware
//
#include <hip/hip_runtime.h>
#include <math.h>

#define B_    64
#define T_    128
#define OUT_  64
#define E_    256
#define H_    512
#define H2_   1024
#define H3_   1536
#define H4_   2048
#define VOUT_ 20000

typedef __attribute__((ext_vector_type(16))) __bf16 v16bf;
typedef __attribute__((ext_vector_type(8)))  __bf16 v8bf;
typedef __attribute__((ext_vector_type(8)))  float  v8f;

union V16U { v16bf v; v8bf h[2]; };

__device__ __forceinline__ float sigf(float x) { return 1.0f / (1.0f + __expf(-x)); }

// ---------------------------------------------------------------------------
// f32 -> bf16 conversion
__global__ void k_f2bf(const float* __restrict__ s, __bf16* __restrict__ d, int n) {
  int i = blockIdx.x * blockDim.x + threadIdx.x;
  if (i < n) d[i] = (__bf16)s[i];
}

// Embedding gathers (output layout (T,B,E) / (OUT,B,E), bf16)
__global__ void k_embed_fwd(const int* __restrict__ x, const float* __restrict__ emb,
                            __bf16* __restrict__ dst) {
  int i = blockIdx.x * blockDim.x + threadIdx.x;   // over T*B*E
  int e  = i & (E_ - 1);
  int tb = i / E_;
  int b  = tb & (B_ - 1);
  int t  = tb / B_;
  int tok = x[b * T_ + t];
  dst[i] = (__bf16)emb[(size_t)tok * E_ + e];
}

__global__ void k_embed_bwd(const int* __restrict__ x, const int* __restrict__ lens,
                            const float* __restrict__ emb, __bf16* __restrict__ dst) {
  int i = blockIdx.x * blockDim.x + threadIdx.x;
  int e  = i & (E_ - 1);
  int tb = i / E_;
  int b  = tb & (B_ - 1);
  int t  = tb / B_;
  int pos = lens[b] - 1 - t;           // clip(len-1-t, 0)
  if (pos < 0) pos = 0;
  int tok = x[b * T_ + pos];
  dst[i] = (__bf16)emb[(size_t)tok * E_ + e];
}

__global__ void k_embed_dec(const int* __restrict__ y, const float* __restrict__ emb,
                            __bf16* __restrict__ dst) {
  int i = blockIdx.x * blockDim.x + threadIdx.x;   // over OUT*B*E
  int e  = i & (E_ - 1);
  int tb = i / E_;
  int b  = tb & (B_ - 1);
  int s  = tb / B_;
  int tok = (s == 0) ? 1 /*SOS*/ : y[b * OUT_ + (s - 1)];
  dst[i] = (__bf16)emb[(size_t)tok * E_ + e];
}

// ---------------------------------------------------------------------------
// Fused LSTM cell step: gates = x@Wi^T + h@Wh^T + b  (bf16 WMMA, f32 accum),
// then elementwise i/f/g/o nonlinearity + c/h update + masked enc write.
// grid = (H/16, B/16), block = 128 (4 wave32, one wave per gate).
__global__ void k_lstm_cell(const __bf16* __restrict__ x_t,   // B x E
                            const __bf16* __restrict__ Wi,    // 4H x E
                            const __bf16* __restrict__ Wh,    // 4H x H
                            const float*  __restrict__ bias,  // 4H
                            const __bf16* __restrict__ h_prev,// B x h_ld (cols h_off..h_off+H)
                            __bf16*       __restrict__ h_next,
                            float*        __restrict__ h_f32, // B x H or null
                            float*        __restrict__ c_state,// B x c_ld
                            const int*    __restrict__ lens,  // null => always update
                            float*        __restrict__ enc_out,// (T,B,2H) or null
                            int enc_mode, int t,
                            int h_ld, int c_ld, int h_off) {
  __shared__ float G[4][16][16];
  const int tid  = threadIdx.x;
  const int wave = tid >> 5;
  const int lane = tid & 31;
  const int half = lane >> 4;
  const int l    = lane & 15;
  const int j0 = blockIdx.x * 16;          // hidden column tile
  const int m0 = blockIdx.y * 16;          // batch row tile
  const int n0 = wave * H_ + j0;           // gate-space column base

  v8f acc;
  {
    float bv = bias[n0 + l];
#pragma unroll
    for (int r = 0; r < 8; ++r) acc[r] = bv;
  }
  // x @ Wi^T
  for (int kb = 0; kb < E_; kb += 32) {
    V16U a, w;
    a.h[0] = *(const v8bf*)(x_t + (m0 + l) * E_ + kb + half * 8);
    a.h[1] = *(const v8bf*)(x_t + (m0 + l) * E_ + kb + 16 + half * 8);
    w.h[0] = *(const v8bf*)(Wi + (size_t)(n0 + l) * E_ + kb + half * 8);
    w.h[1] = *(const v8bf*)(Wi + (size_t)(n0 + l) * E_ + kb + 16 + half * 8);
    acc = __builtin_amdgcn_wmma_f32_16x16x32_bf16(false, a.v, false, w.v, (short)0, acc, false, false);
  }
  // h @ Wh^T
  {
    const __bf16* hp = h_prev + h_off;
    for (int kb = 0; kb < H_; kb += 32) {
      V16U a, w;
      a.h[0] = *(const v8bf*)(hp + (m0 + l) * h_ld + kb + half * 8);
      a.h[1] = *(const v8bf*)(hp + (m0 + l) * h_ld + kb + 16 + half * 8);
      w.h[0] = *(const v8bf*)(Wh + (size_t)(n0 + l) * H_ + kb + half * 8);
      w.h[1] = *(const v8bf*)(Wh + (size_t)(n0 + l) * H_ + kb + 16 + half * 8);
      acc = __builtin_amdgcn_wmma_f32_16x16x32_bf16(false, a.v, false, w.v, (short)0, acc, false, false);
    }
  }
#pragma unroll
  for (int r = 0; r < 8; ++r) G[wave][r + half * 8][l] = acc[r];
  __syncthreads();

  for (int e = tid; e < 256; e += 128) {
    int m = e >> 4, j = e & 15;
    int b = m0 + m, col = j0 + j;
    int len = 0; bool upd = true;
    if (lens) { len = lens[b]; upd = (t < len); }
    int hIdx = b * h_ld + h_off + col;
    if (upd) {
      float iv = sigf(G[0][m][j]);
      float fv = sigf(G[1][m][j]);
      float gv = tanhf(G[2][m][j]);
      float ov = sigf(G[3][m][j]);
      int cIdx = b * c_ld + h_off + col;
      float c2 = fv * c_state[cIdx] + iv * gv;
      float h2 = ov * tanhf(c2);
      c_state[cIdx] = c2;
      h_next[hIdx] = (__bf16)h2;
      if (h_f32) h_f32[b * H_ + col] = h2;
      if (enc_out) {
        int pos  = (enc_mode == 1) ? t : (len - 1 - t);
        int coff = (enc_mode == 1) ? col : (H_ + col);
        enc_out[((size_t)pos * B_ + b) * H2_ + coff] = h2;
      }
    } else {
      h_next[hIdx] = h_prev[hIdx];   // carry old state into new buffer
    }
  }
}

// ---------------------------------------------------------------------------
// Register-blocked bf16 WMMA GEMM for M == 64:
//   out(64 x N) = A(64 x K) @ W^T (W is N x K row-major) + bias.
// Each wave owns a full 64-row strip of 32 columns: 4 m-tiles x 2 n-tiles =
// 8 f32 accumulators. Per K-step: 12 b128 loads feed 8 WMMAs (B fragments
// reused across the 4 m-tiles, A fragments across the 2 n-tiles).
__global__ void k_gemm_bf16_m64(const __bf16* __restrict__ A, int lda,
                                const __bf16* __restrict__ W, int ldb,
                                const float* __restrict__ bias,
                                float* __restrict__ outF,
                                __bf16* __restrict__ outBF, int ldd,
                                int N, int K) {
  const int tid  = threadIdx.x;
  const int wave = tid >> 5;
  const int lane = tid & 31;
  const int half = lane >> 4;
  const int l    = lane & 15;
  const int jobs = N >> 5;                 // pairs of 16-wide n-tiles
  const int job  = blockIdx.x * 8 + wave;
  if (job >= jobs) return;                 // wave-uniform exit
  const int tn0 = job << 5;                // 32-column base

  v8f acc[8];                              // [mi*2 + ni]
#pragma unroll
  for (int ni = 0; ni < 2; ++ni) {
    float bv = bias ? bias[tn0 + ni * 16 + l] : 0.0f;
#pragma unroll
    for (int mi = 0; mi < 4; ++mi)
#pragma unroll
      for (int r = 0; r < 8; ++r) acc[mi * 2 + ni][r] = bv;
  }

  for (int kb = 0; kb < K; kb += 32) {
    V16U w[2];
#pragma unroll
    for (int ni = 0; ni < 2; ++ni) {
      const __bf16* wr = W + (size_t)(tn0 + ni * 16 + l) * ldb + kb + half * 8;
      w[ni].h[0] = *(const v8bf*)(wr);
      w[ni].h[1] = *(const v8bf*)(wr + 16);
    }
    V16U a[4];
#pragma unroll
    for (int mi = 0; mi < 4; ++mi) {
      const __bf16* ar = A + (size_t)(mi * 16 + l) * lda + kb + half * 8;
      a[mi].h[0] = *(const v8bf*)(ar);
      a[mi].h[1] = *(const v8bf*)(ar + 16);
    }
#pragma unroll
    for (int mi = 0; mi < 4; ++mi) {
#pragma unroll
      for (int ni = 0; ni < 2; ++ni) {
        acc[mi * 2 + ni] = __builtin_amdgcn_wmma_f32_16x16x32_bf16(
            false, a[mi].v, false, w[ni].v, (short)0, acc[mi * 2 + ni], false, false);
      }
    }
  }

#pragma unroll
  for (int mi = 0; mi < 4; ++mi) {
#pragma unroll
    for (int ni = 0; ni < 2; ++ni) {
#pragma unroll
      for (int r = 0; r < 8; ++r) {
        int row = mi * 16 + r + half * 8;
        int col = tn0 + ni * 16 + l;
        if (outF)  outF[(size_t)row * ldd + col]  = acc[mi * 2 + ni][r];
        if (outBF) outBF[(size_t)row * ldd + col] = (__bf16)acc[mi * 2 + ni][r];
      }
    }
  }
}

// ---------------------------------------------------------------------------
// Attention for one decoder step. One block per batch element, 128 threads.
// q = h2 @ W_V^T + b_V ; e[t] = dot(q, enc[t,b,:]) ; softmax over t ;
// cvec uses the reference's reshape quirk: enc_flat[b,t,:] = flat row b*T + t.
// Writes Z[b] = [bf16(cvec), bf16(h2)].
__global__ void k_attention(const float* __restrict__ h32,
                            const float* __restrict__ W_V,
                            const float* __restrict__ b_V,
                            const float* __restrict__ enc,
                            const __bf16* __restrict__ hbf,
                            __bf16* __restrict__ Z) {
  __shared__ float q[H2_];
  __shared__ float sA[T_];
  __shared__ float red[128];
  const int tid = threadIdx.x;
  const int b   = blockIdx.x;

  const float* hv = h32 + b * H_;
#pragma unroll 1
  for (int r = 0; r < 8; ++r) {
    int j = r * 128 + tid;
    float a = b_V[j];
    const float* wv = W_V + (size_t)j * H_;
    for (int k = 0; k < H_; ++k) a = fmaf(hv[k], wv[k], a);
    q[j] = a;
  }
  __syncthreads();

  float ev = 0.0f;
  {
    const float* er = enc + ((size_t)tid * B_ + b) * H2_;   // enc[t=tid, b, :]
    for (int k = 0; k < H2_; ++k) ev = fmaf(q[k], er[k], ev);
  }
  red[tid] = ev;
  __syncthreads();
  for (int s = 64; s > 0; s >>= 1) {
    if (tid < s) red[tid] = fmaxf(red[tid], red[tid + s]);
    __syncthreads();
  }
  float mx = red[0];
  __syncthreads();
  float pv = __expf(ev - mx);
  red[tid] = pv;
  __syncthreads();
  for (int s = 64; s > 0; s >>= 1) {
    if (tid < s) red[tid] += red[tid + s];
    __syncthreads();
  }
  float inv = 1.0f / red[0];
  sA[tid] = pv * inv;
  __syncthreads();

#pragma unroll 1
  for (int r = 0; r < 8; ++r) {
    int j = r * 128 + tid;
    float a = 0.0f;
    for (int t = 0; t < T_; ++t)
      a = fmaf(sA[t], enc[((size_t)b * T_ + t) * H2_ + j], a);   // reshape quirk
    Z[(size_t)b * H3_ + j] = (__bf16)a;
  }
  for (int j = tid; j < H_; j += 128)
    Z[(size_t)b * H3_ + H2_ + j] = hbf[b * H_ + j];
}

// ---------------------------------------------------------------------------
extern "C" void kernel_launch(void* const* d_in, const int* in_sizes, int n_in,
                              void* d_out, int out_size, void* d_ws, size_t ws_size,
                              hipStream_t stream) {
  (void)in_sizes; (void)n_in; (void)out_size; (void)ws_size;
  const int*   x       = (const int*)d_in[0];
  const int*   lens    = (const int*)d_in[1];
  const int*   y       = (const int*)d_in[2];
  const float* emb_in  = (const float*)d_in[4];
  const float* emb_out = (const float*)d_in[5];
  const float* W_ih_f  = (const float*)d_in[6];
  const float* W_hh_f  = (const float*)d_in[7];
  const float* b_f     = (const float*)d_in[8];
  const float* W_ih_b  = (const float*)d_in[9];
  const float* W_hh_b  = (const float*)d_in[10];
  const float* b_b     = (const float*)d_in[11];
  const float* W_rh    = (const float*)d_in[12];
  const float* b_rh    = (const float*)d_in[13];
  const float* W_rc    = (const float*)d_in[14];
  const float* b_rc    = (const float*)d_in[15];
  const float* W_ih_d  = (const float*)d_in[16];
  const float* W_hh_d  = (const float*)d_in[17];
  const float* b_d     = (const float*)d_in[18];
  const float* W_V     = (const float*)d_in[19];
  const float* b_V     = (const float*)d_in[20];
  const float* W_fc    = (const float*)d_in[21];
  const float* b_fc    = (const float*)d_in[22];
  float* out = (float*)d_out;

  char* ws = (char*)d_ws;
  size_t off = 0;
  auto alloc = [&](size_t bytes) -> char* {
    char* p = ws + off;
    off = (off + bytes + 255) & ~(size_t)255;
    return p;
  };

  __bf16* Wi_f  = (__bf16*)alloc((size_t)H4_ * E_ * 2);
  __bf16* Wh_f  = (__bf16*)alloc((size_t)H4_ * H_ * 2);
  __bf16* Wi_b  = (__bf16*)alloc((size_t)H4_ * E_ * 2);
  __bf16* Wh_b  = (__bf16*)alloc((size_t)H4_ * H_ * 2);
  __bf16* Wi_d  = (__bf16*)alloc((size_t)H4_ * E_ * 2);
  __bf16* Wh_d  = (__bf16*)alloc((size_t)H4_ * H_ * 2);
  __bf16* Wrh   = (__bf16*)alloc((size_t)H_ * H2_ * 2);
  __bf16* Wrc   = (__bf16*)alloc((size_t)H_ * H2_ * 2);
  __bf16* Wfc   = (__bf16*)alloc((size_t)VOUT_ * H3_ * 2);   // 61.4 MB: L2-resident
  __bf16* xsf   = (__bf16*)alloc((size_t)T_ * B_ * E_ * 2);
  __bf16* xsb   = (__bf16*)alloc((size_t)T_ * B_ * E_ * 2);
  __bf16* xde   = (__bf16*)alloc((size_t)OUT_ * B_ * E_ * 2);
  float*  enc   = (float*)alloc((size_t)T_ * B_ * H2_ * 4);  // 33.5 MB: L2-resident
  __bf16* hcat0 = (__bf16*)alloc((size_t)B_ * H2_ * 2);      // [hf | hb] double-buffered
  __bf16* hcat1 = (__bf16*)alloc((size_t)B_ * H2_ * 2);
  float*  ccat  = (float*)alloc((size_t)B_ * H2_ * 4);       // [cf | cb] f32
  __bf16* ccatb = (__bf16*)alloc((size_t)B_ * H2_ * 2);
  __bf16* hdec0 = (__bf16*)alloc((size_t)B_ * H_ * 2);
  __bf16* hdec1 = (__bf16*)alloc((size_t)B_ * H_ * 2);
  float*  hd32  = (float*)alloc((size_t)B_ * H_ * 4);
  float*  cdec  = (float*)alloc((size_t)B_ * H_ * 4);
  __bf16* Z     = (__bf16*)alloc((size_t)B_ * H3_ * 2);

  auto cv = [&](const float* s, __bf16* d, size_t n) {
    k_f2bf<<<dim3((unsigned)((n + 255) / 256)), dim3(256), 0, stream>>>(s, d, (int)n);
  };
  cv(W_ih_f, Wi_f, (size_t)H4_ * E_);
  cv(W_hh_f, Wh_f, (size_t)H4_ * H_);
  cv(W_ih_b, Wi_b, (size_t)H4_ * E_);
  cv(W_hh_b, Wh_b, (size_t)H4_ * H_);
  cv(W_ih_d, Wi_d, (size_t)H4_ * E_);
  cv(W_hh_d, Wh_d, (size_t)H4_ * H_);
  cv(W_rh,   Wrh,  (size_t)H_ * H2_);
  cv(W_rc,   Wrc,  (size_t)H_ * H2_);
  cv(W_fc,   Wfc,  (size_t)VOUT_ * H3_);

  k_embed_fwd<<<dim3((T_ * B_ * E_) / 256), dim3(256), 0, stream>>>(x, emb_in, xsf);
  k_embed_bwd<<<dim3((T_ * B_ * E_) / 256), dim3(256), 0, stream>>>(x, lens, emb_in, xsb);
  k_embed_dec<<<dim3((OUT_ * B_ * E_) / 256), dim3(256), 0, stream>>>(y, emb_out, xde);

  hipMemsetAsync(enc,   0, (size_t)T_ * B_ * H2_ * 4, stream);
  hipMemsetAsync(hcat0, 0, (size_t)B_ * H2_ * 2, stream);
  hipMemsetAsync(hcat1, 0, (size_t)B_ * H2_ * 2, stream);
  hipMemsetAsync(ccat,  0, (size_t)B_ * H2_ * 4, stream);

  dim3 cg(H_ / 16, B_ / 16);   // (32, 4) blocks x 128 threads

  // Encoder forward (h in cols 0..H-1 of hcat, c in cols 0..H-1 of ccat)
  for (int t = 0; t < T_; ++t) {
    __bf16* hp = (t & 1) ? hcat1 : hcat0;
    __bf16* hn = (t & 1) ? hcat0 : hcat1;
    k_lstm_cell<<<cg, 128, 0, stream>>>(xsf + (size_t)t * B_ * E_, Wi_f, Wh_f, b_f,
                                        hp, hn, nullptr, ccat, lens, enc,
                                        /*mode=*/1, t, H2_, H2_, /*off=*/0);
  }
  // Encoder backward (cols H..2H-1); T even => both directions end in hcat0
  for (int t = 0; t < T_; ++t) {
    __bf16* hp = (t & 1) ? hcat1 : hcat0;
    __bf16* hn = (t & 1) ? hcat0 : hcat1;
    k_lstm_cell<<<cg, 128, 0, stream>>>(xsb + (size_t)t * B_ * E_, Wi_b, Wh_b, b_b,
                                        hp, hn, nullptr, ccat, lens, enc,
                                        /*mode=*/2, t, H2_, H2_, /*off=*/H_);
  }

  // h0 = [hf|hb] @ W_rh^T + b_rh ; c0 = [cf|cb] @ W_rc^T + b_rc
  cv(ccat, ccatb, (size_t)B_ * H2_);
  k_gemm_bf16_m64<<<dim3(2), dim3(256), 0, stream>>>(hcat0, H2_, Wrh, H2_, b_rh,
                                                     nullptr, hdec0, H_, H_, H2_);
  k_gemm_bf16_m64<<<dim3(2), dim3(256), 0, stream>>>(ccatb, H2_, Wrc, H2_, b_rc,
                                                     cdec, nullptr, H_, H_, H2_);

  // Decoder: cell -> attention -> logits per step
  for (int s = 0; s < OUT_; ++s) {
    __bf16* hp = (s & 1) ? hdec1 : hdec0;
    __bf16* hn = (s & 1) ? hdec0 : hdec1;
    k_lstm_cell<<<cg, 128, 0, stream>>>(xde + (size_t)s * B_ * E_, Wi_d, Wh_d, b_d,
                                        hp, hn, hd32, cdec, nullptr, nullptr,
                                        /*mode=*/0, s, H_, H_, /*off=*/0);
    k_attention<<<dim3(B_), dim3(128), 0, stream>>>(hd32, W_V, b_V, enc, hn, Z);
    // 20000/32 = 625 wave-jobs, 8 waves/block -> 79 blocks
    k_gemm_bf16_m64<<<dim3(79), dim3(256), 0, stream>>>(Z, H3_, Wfc, H3_, b_fc,
                                                        out + (size_t)s * B_ * VOUT_,
                                                        nullptr, VOUT_, VOUT_, H3_);
  }
}
